// OptimizedMoE_31086973288516
// MI455X (gfx1250) — matrix operand order repaired and checked by reading, not compiled, not run on hardware
//
#include <hip/hip_runtime.h>
#include <hip/hip_bf16.h>
#include <stdint.h>

#define T_TOK 4096
#define DIM   1024
#define NEXP  8

#define MT 128      // M tile (tokens)
#define NT 128      // N tile (output features)
#define KC 32       // K chunk per WMMA step
#define LDPAD 40    // padded LDS row stride (bf16 elems, 80B)

typedef __attribute__((ext_vector_type(16))) __bf16 v16bf;
typedef __attribute__((ext_vector_type(8)))  __bf16 v8bf;
typedef __attribute__((ext_vector_type(8)))  float  v8f;
typedef __attribute__((ext_vector_type(4)))  float  v4f;
typedef __attribute__((ext_vector_type(4)))  int    v4i;

// address-space-qualified pointer types for the async builtin
typedef __attribute__((address_space(1))) v4i* as1_v4i_p;
typedef __attribute__((address_space(3))) v4i* as3_v4i_p;

// ---------------- workspace layout ----------------
// counts : int[NEXP]               @ 0
// gate   : float[T_TOK]            @ 256
// list   : int[NEXP][T_TOK]        @ 16640
// xb     : bf16[T_TOK*DIM]         @ 147712   (8.39 MB)
// wb     : bf16[NEXP*DIM*DIM]      @ 8536320  (16.78 MB)
#define WS_GATE_OFF  256
#define WS_LIST_OFF  (WS_GATE_OFF + T_TOK * 4)
#define WS_XB_OFF    ((WS_LIST_OFF + NEXP * T_TOK * 4 + 255) & ~255)
#define WS_WB_OFF    (WS_XB_OFF + T_TOK * DIM * 2)

// ---------------- async global->LDS helpers (CDNA5) ----------------
__device__ __forceinline__ void async_copy16(void* lds_dst, const void* gsrc) {
    // global source: full 64-bit flat address; LDS dest: flat[31:0] == LDS addr
    as1_v4i_p g = (as1_v4i_p)(uintptr_t)gsrc;
    as3_v4i_p l = (as3_v4i_p)(unsigned)(uintptr_t)lds_dst;
    __builtin_amdgcn_global_load_async_to_lds_b128(g, l, 0, 0);
}

__device__ __forceinline__ void wait_async_le4() {
    __builtin_amdgcn_s_wait_asynccnt(4);
}
__device__ __forceinline__ void wait_async_le0() {
    __builtin_amdgcn_s_wait_asynccnt(0);
}

// ---------------- kernels ----------------
__global__ __launch_bounds__(32) void moe_init_counts(int* __restrict__ counts) {
    if (threadIdx.x < NEXP) counts[threadIdx.x] = 0;
}

// One-shot fp32 -> bf16 conversion of x and expert_w into workspace.
__global__ __launch_bounds__(256) void moe_convert_bf16(
    const float* __restrict__ xs, __bf16* __restrict__ xd, long xn8,
    const float* __restrict__ wsrc, __bf16* __restrict__ wd, long wn8)
{
    long i = (long)blockIdx.x * blockDim.x + threadIdx.x;
    const float* __restrict__ s;
    __bf16* __restrict__ d;
    if (i < xn8) { s = xs; d = xd; }
    else {
        i -= xn8;
        if (i >= wn8) return;
        s = wsrc; d = wd;
    }
    const v4f a = *(const v4f*)(s + i * 8);
    const v4f b = *(const v4f*)(s + i * 8 + 4);
    v8bf o;
    o[0] = (__bf16)a.x; o[1] = (__bf16)a.y; o[2] = (__bf16)a.z; o[3] = (__bf16)a.w;
    o[4] = (__bf16)b.x; o[5] = (__bf16)b.y; o[6] = (__bf16)b.z; o[7] = (__bf16)b.w;
    *(v8bf*)(d + i * 8) = o;
}

// One wave32 per token: router logits, softmax gate, top-1 expert, bucket append.
__global__ __launch_bounds__(256) void moe_router(
    const float* __restrict__ x, const float* __restrict__ rw,
    const float* __restrict__ rb,
    int* __restrict__ counts, float* __restrict__ gate, int* __restrict__ list)
{
    const int wave = threadIdx.x >> 5;
    const int lane = threadIdx.x & 31;
    const int t    = blockIdx.x * 8 + wave;
    if (t >= T_TOK) return;

    const float* __restrict__ xr = x + (size_t)t * DIM;

    float acc[NEXP];
#pragma unroll
    for (int e = 0; e < NEXP; ++e) acc[e] = 0.f;

    for (int k = lane; k < DIM; k += 32) {
        const float xv = xr[k];
#pragma unroll
        for (int e = 0; e < NEXP; ++e)
            acc[e] = __builtin_fmaf(xv, rw[e * DIM + k], acc[e]);
    }
#pragma unroll
    for (int e = 0; e < NEXP; ++e) {
        float v = acc[e];
#pragma unroll
        for (int off = 16; off > 0; off >>= 1) v += __shfl_xor(v, off, 32);
        acc[e] = v;
    }

    if (lane == 0) {
        float l[NEXP];
        float m = -1e30f; int best = 0;
#pragma unroll
        for (int e = 0; e < NEXP; ++e) {
            l[e] = acc[e] + rb[e];
            if (l[e] > m) { m = l[e]; best = e; }
        }
        float s = 0.f;
#pragma unroll
        for (int e = 0; e < NEXP; ++e) s += __expf(l[e] - m);
        gate[t] = 1.0f / s;                       // exp(l[best]-m) == 1
        const int pos = atomicAdd(&counts[best], 1);
        list[best * T_TOK + pos] = t;
    }
}

// Grouped expert GEMM on pre-converted bf16 data.
// 128x128 tile, async double-buffered LDS staging, 8x v_wmma_f32_16x16x32_bf16.
// grid = (DIM/NT, NEXP * (T_TOK/MT)), block = 256 (8 waves).
__global__ __launch_bounds__(256) void moe_expert_gemm(
    const __bf16* __restrict__ xb, const __bf16* __restrict__ wb,
    const float* __restrict__ eb,
    const int* __restrict__ counts, const float* __restrict__ gate,
    const int* __restrict__ list, float* __restrict__ out)
{
    __shared__ __bf16 As[2][MT * LDPAD];
    __shared__ __bf16 Bs[2][NT * LDPAD];

    const int e   = blockIdx.y >> 5;         // T_TOK/MT == 32 m-tiles per expert
    const int mt  = blockIdx.y & 31;
    const int cnt = counts[e];
    const int m_base = mt * MT;
    if (m_base >= cnt) return;
    const int rows = min(MT, cnt - m_base);
    const int n0   = blockIdx.x * NT;

    const int*    __restrict__ tl = list + e * T_TOK + m_base;
    const __bf16* __restrict__ W  = wb + (size_t)e * DIM * DIM;

    const int tid  = threadIdx.x;
    const int lane = tid & 31;
    const int wave = tid >> 5;

    // staging map: 2 threads per row, 16 bf16 elems (2x b128) each per chunk
    const int lr = tid >> 1;                  // 0..127
    const int lh = (tid & 1) * 16;            // elem offset 0 or 16 in K chunk
    const int atok = tl[min(lr, rows - 1)];   // clamp tail rows (masked at store)
    const __bf16* __restrict__ arow = xb + (size_t)atok * DIM;
    const __bf16* __restrict__ brow = W + (size_t)(n0 + lr) * DIM;
    __bf16* const adst0 = &As[0][lr * LDPAD + lh];
    __bf16* const adst1 = &As[1][lr * LDPAD + lh];
    __bf16* const bdst0 = &Bs[0][lr * LDPAD + lh];
    __bf16* const bdst1 = &Bs[1][lr * LDPAD + lh];

    v8f acc[8] = {};

    const int khalf = lane >> 4;              // lanes 16..31 hold the +8/+16 K halves
    const int ml    = lane & 15;

    // issue 4 async b128 copies for chunk at k0 into buffer `b`
    auto issue = [&](int k0, int b) {
        __bf16* ad = b ? adst1 : adst0;
        __bf16* bd = b ? bdst1 : bdst0;
        async_copy16(ad,     arow + k0 + lh);
        async_copy16(ad + 8, arow + k0 + lh + 8);
        async_copy16(bd,     brow + k0 + lh);
        async_copy16(bd + 8, brow + k0 + lh + 8);
    };

    issue(0, 0);
    int buf = 0;
    for (int k0 = 0; k0 < DIM; k0 += KC) {
        const bool hasNext = (k0 + KC) < DIM;
        if (hasNext) {
            issue(k0 + KC, buf ^ 1);
            wait_async_le4();                 // current chunk's 4 copies are done
        } else {
            wait_async_le0();
        }
        __syncthreads();                      // all waves' data visible in LDS

        const __bf16* ap0 = &As[buf][(wave * 16 + ml) * LDPAD + khalf * 8];
        union { v16bf v; v8bf h[2]; } afrag;
        afrag.h[0] = *(const v8bf*)ap0;       // K = khalf*8 .. +7
        afrag.h[1] = *(const v8bf*)(ap0 + 16);// K = 16+khalf*8 .. +7
#pragma unroll
        for (int f = 0; f < 8; ++f) {
            const __bf16* bp = &Bs[buf][(f * 16 + ml) * LDPAD + khalf * 8];
            union { v16bf v; v8bf h[2]; } bfrag;
            bfrag.h[0] = *(const v8bf*)bp;    // K = khalf*8 .. +7, N = ml
            bfrag.h[1] = *(const v8bf*)(bp + 16);
            acc[f] = __builtin_amdgcn_wmma_f32_16x16x32_bf16(
                false, afrag.v, false, bfrag.v, (short)0, acc[f], false, false);
        }
        __syncthreads();                      // done reading buf before it refills
        buf ^= 1;
    }

    // epilogue: out[t, n] = (acc + bias[n]) * gate[t]
    const int hi = lane >> 4;
#pragma unroll
    for (int f = 0; f < 8; ++f) {
        const float bias = eb[e * DIM + n0 + f * 16 + ml];
#pragma unroll
        for (int v = 0; v < 8; ++v) {
            const int r = wave * 16 + v + 8 * hi;   // C layout: VGPR v -> M=v (+8 hi lanes)
            if (r < rows) {
                const int t = tl[r];
                out[(size_t)t * DIM + n0 + f * 16 + ml] = (acc[f][v] + bias) * gate[t];
            }
        }
    }
}

extern "C" void kernel_launch(void* const* d_in, const int* in_sizes, int n_in,
                              void* d_out, int out_size, void* d_ws, size_t ws_size,
                              hipStream_t stream) {
    const float* x   = (const float*)d_in[0];   // [2,2048,1024]
    const float* rw  = (const float*)d_in[1];   // [8,1024]
    const float* rb  = (const float*)d_in[2];   // [8]
    const float* ew  = (const float*)d_in[3];   // [8,1024,1024]
    const float* eb  = (const float*)d_in[4];   // [8,1024]
    float*       out = (float*)d_out;

    int*    counts = (int*)d_ws;
    float*  gate   = (float*)((char*)d_ws + WS_GATE_OFF);
    int*    list   = (int*)((char*)d_ws + WS_LIST_OFF);
    __bf16* xbuf   = (__bf16*)((char*)d_ws + WS_XB_OFF);
    __bf16* wbuf   = (__bf16*)((char*)d_ws + WS_WB_OFF);

    const long xn8 = (long)T_TOK * DIM / 8;           // 524288 threads for x
    const long wn8 = (long)NEXP * DIM * DIM / 8;      // 1048576 threads for W
    const int  cvt_blocks = (int)((xn8 + wn8 + 255) / 256);   // 6144

    moe_init_counts<<<1, 32, 0, stream>>>(counts);
    moe_convert_bf16<<<cvt_blocks, 256, 0, stream>>>(x, xbuf, xn8, ew, wbuf, wn8);
    moe_router<<<T_TOK / 8, 256, 0, stream>>>(x, rw, rb, counts, gate, list);
    dim3 grid(DIM / NT, NEXP * (T_TOK / MT));
    moe_expert_gemm<<<grid, 256, 0, stream>>>(xbuf, wbuf, eb, counts, gate, list, out);
}
